// LowPrecisionLeNet_85590108274949
// MI455X (gfx1250) — matrix operand order repaired and checked by reading, not compiled, not run on hardware
//
#include <hip/hip_runtime.h>
#include <hip/hip_bf16.h>

typedef __attribute__((ext_vector_type(16))) _Float16 v16h;
typedef __attribute__((ext_vector_type(8)))  _Float16 v8h;
typedef __attribute__((ext_vector_type(8)))  float    v8f;
typedef __attribute__((ext_vector_type(4)))  unsigned v4u;
typedef __attribute__((ext_vector_type(8)))  int      v8i;
typedef __attribute__((ext_vector_type(4)))  int      v4i;

#define LDA 72   // 64 + 8 f16 pad (row stride 144B: 16B-aligned, bank-spread)
#define LDB 72   // matches TDM pad_interval=32 DW, pad_amount=4 DW

#if defined(__gfx1250__) && __has_builtin(__builtin_amdgcn_tensor_load_to_lds) && \
    __has_builtin(__builtin_amdgcn_s_wait_tensorcnt)
#define USE_TDM 1
#else
#define USE_TDM 0
#endif

// ---------------------------------------------------------------------------
// Small utility kernels
// ---------------------------------------------------------------------------
__global__ void k_zero16(unsigned* sc) {
    if (threadIdx.x < 16) sc[threadIdx.x] = 0u;
}

__global__ void k_absmax(const float* __restrict__ x, long long n,
                         unsigned* __restrict__ out) {
    long long i = (long long)blockIdx.x * blockDim.x + threadIdx.x;
    long long stride = (long long)gridDim.x * blockDim.x;
    float m = 0.f;
    for (; i < n; i += stride) m = fmaxf(m, fabsf(x[i]));
    atomicMax(out, __float_as_uint(m));   // valid: non-negative floats
}

// signed 4-bit fake-quant of input: q = clip(round(x/s), -8, 7), s = max|x|/7
__global__ void k_quant_s4(const float* __restrict__ x, long long n,
                           const unsigned* __restrict__ maxbits,
                           _Float16* __restrict__ q) {
    float s = fmaxf(__uint_as_float(*maxbits), 1e-8f) / 7.f;
    long long i = (long long)blockIdx.x * blockDim.x + threadIdx.x;
    long long stride = (long long)gridDim.x * blockDim.x;
    for (; i < n; i += stride) {
        float v = rintf(x[i] / s);
        v = fminf(fmaxf(v, -8.f), 7.f);
        q[i] = (_Float16)v;
    }
}

// unsigned 4-bit fake-quant of post-ReLU activations: q = clip(round(y/s),0,15)
__global__ void k_quant_u4(const float* __restrict__ y, long long n,
                           const unsigned* __restrict__ maxbits,
                           _Float16* __restrict__ q) {
    float s = fmaxf(__uint_as_float(*maxbits), 1e-8f) / 15.f;
    long long i = (long long)blockIdx.x * blockDim.x + threadIdx.x;
    long long stride = (long long)gridDim.x * blockDim.x;
    for (; i < n; i += stride) {
        float v = rintf(y[i] / s);
        v = fminf(fmaxf(v, 0.f), 15.f);
        q[i] = (_Float16)v;
    }
}

// 3-bit narrow-symmetric weight quant, single block: reduce max|w|, then write
// integer-valued f16 weights, zero-padded to [Npad][Kpad].
__global__ void k_quant_w(const float* __restrict__ w, int Nr, int Npad,
                          int Kr, int Kpad,
                          unsigned* __restrict__ maxbits,
                          _Float16* __restrict__ wq, float qmax) {
    __shared__ float red[256];
    const int tid = threadIdx.x;
    float m = 0.f;
    for (int i = tid; i < Nr * Kr; i += 256) m = fmaxf(m, fabsf(w[i]));
    red[tid] = m;
    __syncthreads();
    for (int s2 = 128; s2 > 0; s2 >>= 1) {
        if (tid < s2) red[tid] = fmaxf(red[tid], red[tid + s2]);
        __syncthreads();
    }
    if (tid == 0) *maxbits = __float_as_uint(red[0]);
    float s = fmaxf(red[0], 1e-8f) / qmax;
    for (int idx = tid; idx < Npad * Kpad; idx += 256) {
        int n = idx / Kpad;
        int k = idx - n * Kpad;
        float q = 0.f;
        if (n < Nr && k < Kr) {
            q = rintf(w[n * Kr + k] / s);
            q = fminf(fmaxf(q, -qmax), qmax);
        }
        wq[idx] = (_Float16)q;
    }
}

// 2x2 maxpool on integer-valued f16 activations (scale>0 commutes with max)
__global__ void k_maxpool(const _Float16* __restrict__ in,
                          _Float16* __restrict__ out,
                          int BC, int H, int W, int OH, int OW) {
    long long n = (long long)BC * OH * OW;
    long long i = (long long)blockIdx.x * blockDim.x + threadIdx.x;
    if (i >= n) return;
    int ow = (int)(i % OW);
    long long t = i / OW;
    int oh = (int)(t % OH);
    long long bc = t / OH;
    const _Float16* p = in + (bc * H + oh * 2) * W + ow * 2;
    float a = (float)p[0], b = (float)p[1], c = (float)p[W], d = (float)p[W + 1];
    out[i] = (_Float16)fmaxf(fmaxf(a, b), fmaxf(c, d));
}

// ---------------------------------------------------------------------------
// WMMA implicit-GEMM: C[M,N] = A[M,K] * Wq[N,K]^T  (integer-valued f16 inputs)
// B tile staged LDS via Tensor Data Mover (2D descriptor, padded LDS rows).
// Block = 256 threads = 8 waves; each wave one 16x16 tile; block tile 64x32.
// ---------------------------------------------------------------------------
__global__ __launch_bounds__(256)
void k_gemm_wmma(const _Float16* __restrict__ A,
                 const _Float16* __restrict__ Wq,
                 const float* __restrict__ bias,
                 float* __restrict__ Y,
                 unsigned* __restrict__ outMax,
                 const unsigned* __restrict__ aMaxBits,
                 const unsigned* __restrict__ wMaxBits,
                 float aQmax, float wQmax,
                 int M, int N, int Npad, int Kreal, int Kpad,
                 int conv, int Cin, int H, int W, int OH, int OW,
                 int relu) {
    __shared__ __align__(16) _Float16 As[64 * LDA];
    __shared__ __align__(16) _Float16 Bs[32 * LDB];

    const int tid    = threadIdx.x;
    const int lane   = tid & 31;
    const int wid    = tid >> 5;
    const int mt     = wid & 3;
    const int nt     = wid >> 2;
    const int halfid = lane >> 4;
    const int l16    = lane & 15;

    const int m0 = blockIdx.x * 64;
    const int n0 = blockIdx.y * 32;
    const int OHW = OH * OW;

    const float sA = fmaxf(__uint_as_float(*aMaxBits), 1e-8f) / aQmax;
    const float sW = fmaxf(__uint_as_float(*wMaxBits), 1e-8f) / wQmax;

    v8f acc = {0.f, 0.f, 0.f, 0.f, 0.f, 0.f, 0.f, 0.f};

    // ---- per-thread A-gather coordinates: one row, 16 consecutive K ----
    const int arow  = tid >> 2;            // 0..63
    const int kbase = (tid & 3) * 16;      // 0,16,32,48
    const int am    = m0 + arow;
    const bool mval = am < M;
    int ab = 0, aoy = 0, aox = 0;
    if (conv && mval) {
        ab = am / OHW;
        int rr = am - ab * OHW;
        aoy = rr / OW;
        aox = rr - aoy * OW;
    }

#if USE_TDM
    // ---- constant part of TDM descriptor for B tile (32 x 64 f16) ----
    const unsigned ldsB = (unsigned)(uintptr_t)&Bs[0];
    v8i g1;
    g1[0] = 0x07110000;  // data_size=2B, pad_enable, pad_interval=32DW, pad_amount=4DW
    g1[1] = (int)((Kpad & 0xFFFF) << 16);                        // tensor_dim0 lo
    g1[2] = (int)(((Kpad >> 16) & 0xFFFF) | ((Npad & 0xFFFF) << 16));
    g1[3] = (int)(((Npad >> 16) & 0xFFFF) | (64 << 16));         // tile_dim0=64
    g1[4] = 32;                                                  // tile_dim1=32
    g1[5] = (int)Kpad;                                           // dim0 stride lo
    g1[6] = 0;
    g1[7] = 0;
    const v4i gz4 = {0, 0, 0, 0};
    const v8i gz8 = {0, 0, 0, 0, 0, 0, 0, 0};
#endif

    for (int k0 = 0; k0 < Kpad; k0 += 64) {
        if (k0 + 64 < Kpad)  // warm GL2 for the next weight tile
            __builtin_prefetch(&Wq[(long long)n0 * Kpad + k0 + 64], 0, 0);

        // ---- A chunk: thread writes 16 consecutive K of one row ----
        {
            int k  = k0 + kbase;
            int ic = 0, ky = 0, kx = 0;
            if (conv) {
                ic = k / 9;
                int t = k - ic * 9;
                ky = t / 3;
                kx = t - ky * 3;
            }
            #pragma unroll
            for (int j = 0; j < 16; ++j) {
                _Float16 v = (_Float16)0.f;
                if (mval && k < Kreal) {
                    if (conv)
                        v = A[((long long)(ab * Cin + ic) * H + (aoy + ky)) * W + (aox + kx)];
                    else
                        v = A[(long long)am * Kreal + k];
                }
                As[arow * LDA + kbase + j] = v;
                ++k;
                if (conv) {
                    if (++kx == 3) { kx = 0; if (++ky == 3) { ky = 0; ++ic; } }
                }
            }
        }

#if USE_TDM
        // ---- B chunk via Tensor Data Mover (wave 0 issues, all sync) ----
        if (wid == 0) {
            unsigned long long ga = (unsigned long long)(uintptr_t)Wq +
                ((unsigned long long)n0 * (unsigned)Kpad + (unsigned)k0) * 2ull;
            v4u g0;
            g0[0] = 1u;                                   // count=1 valid descriptor
            g0[1] = ldsB;                                 // LDS dest
            g0[2] = (unsigned)ga;                         // global_addr[31:0]
            g0[3] = (unsigned)((ga >> 32) & 0x1FFFFFFu) | (2u << 30);  // type=2
            __builtin_amdgcn_tensor_load_to_lds(g0, g1, gz4, gz4, gz8, 0);
            __builtin_amdgcn_s_wait_tensorcnt(0);
        }
#else
        // ---- B chunk: thread loads 8 contiguous K of one weight row ----
        {
            int brow = tid >> 3;             // 0..31
            int bk   = (tid & 7) * 8;        // 0..56
            v8h bv = *(const v8h*)&Wq[(long long)(n0 + brow) * Kpad + k0 + bk];
            *(v8h*)&Bs[brow * LDB + bk] = bv;
        }
#endif
        __syncthreads();

        // ---- two 16x16x32 WMMAs per 64-K chunk ----
        #pragma unroll
        for (int ks = 0; ks < 64; ks += 32) {
            const int mrow = mt * 16 + l16;
            const int kbA  = ks + halfid * 16;                 // A: K 0-15 / 16-31 halves
            v8h alo = *(const v8h*)&As[mrow * LDA + kbA];      // K = kbA..kbA+7
            v8h ahi = *(const v8h*)&As[mrow * LDA + kbA + 8];  // K = kbA+8..kbA+15
            v16h afrag = __builtin_shufflevector(alo, ahi,
                0, 1, 2, 3, 4, 5, 6, 7, 8, 9, 10, 11, 12, 13, 14, 15);

            const int nrow = nt * 16 + l16;
            const int kbB  = ks + halfid * 16;                 // B: K 0-15 / 16-31 halves
            v8h blo = *(const v8h*)&Bs[nrow * LDB + kbB];
            v8h bhi = *(const v8h*)&Bs[nrow * LDB + kbB + 8];
            v16h bfrag = __builtin_shufflevector(blo, bhi,
                0, 1, 2, 3, 4, 5, 6, 7, 8, 9, 10, 11, 12, 13, 14, 15);

            acc = __builtin_amdgcn_wmma_f32_16x16x32_f16(
                false, afrag, false, bfrag, (short)0, acc, false, false);
        }
        __syncthreads();
    }

    // ---- epilogue: dequant + bias + relu + store + global max ----
    const float scale = sA * sW;
    const int n = n0 + nt * 16 + l16;
    float lmax = 0.f;
    #pragma unroll
    for (int r = 0; r < 8; ++r) {
        int m = m0 + mt * 16 + halfid * 8 + r;   // C layout: M = r + 8*(lane>=16)
        if (m < M && n < N) {
            float y = acc[r] * scale + (bias ? bias[n] : 0.f);
            if (relu) y = fmaxf(y, 0.f);
            long long idx;
            if (conv) {
                int b  = m / OHW;
                int rr = m - b * OHW;
                idx = ((long long)b * N + n) * OHW + rr;   // NCHW
            } else {
                idx = (long long)m * N + n;
            }
            Y[idx] = y;
            lmax = fmaxf(lmax, y);
        }
    }
    if (outMax) atomicMax(outMax, __float_as_uint(lmax));
}

// ---------------------------------------------------------------------------
// Host-side orchestration
// ---------------------------------------------------------------------------
extern "C" void kernel_launch(void* const* d_in, const int* in_sizes, int n_in,
                              void* d_out, int out_size, void* d_ws, size_t ws_size,
                              hipStream_t stream) {
    (void)n_in; (void)out_size; (void)ws_size;
    const float* x   = (const float*)d_in[0];
    const float* w1  = (const float*)d_in[1];
    const float* b1  = (const float*)d_in[2];
    const float* w2  = (const float*)d_in[3];
    const float* b2  = (const float*)d_in[4];
    const float* w3  = (const float*)d_in[5];
    const float* b3  = (const float*)d_in[6];
    const float* fw1 = (const float*)d_in[7];
    const float* fb1 = (const float*)d_in[8];
    const float* fw2 = (const float*)d_in[9];
    const float* fb2 = (const float*)d_in[10];
    const float* fw3 = (const float*)d_in[11];
    float* out = (float*)d_out;

    const int B = in_sizes[0] / (3 * 32 * 32);

    // bump allocator over workspace
    char* base = (char*)d_ws;
    size_t off = 0;
    auto alloc = [&](size_t bytes) -> void* {
        off = (off + 255) & ~(size_t)255;
        void* p = base + off;
        off += bytes;
        return p;
    };

    unsigned* sc = (unsigned*)alloc(64);   // max-value slots
    // weight buffers zero-padded to [Npad][Kpad] so TDM tiles are in-bounds
    _Float16* w1q = (_Float16*)alloc((size_t)32  * 64   * 2);
    _Float16* w2q = (_Float16*)alloc((size_t)32  * 192  * 2);
    _Float16* w3q = (_Float16*)alloc((size_t)64  * 320  * 2);
    _Float16* f1q = (_Float16*)alloc((size_t)128 * 1024 * 2);
    _Float16* f2q = (_Float16*)alloc((size_t)96  * 128  * 2);
    _Float16* f3q = (_Float16*)alloc((size_t)32  * 128  * 2);

    const size_t nx  = (size_t)B * 3  * 32 * 32;
    const size_t n1  = (size_t)B * 16 * 30 * 30;
    const size_t p1n = (size_t)B * 16 * 15 * 15;
    const size_t n2  = (size_t)B * 32 * 13 * 13;
    const size_t p2n = (size_t)B * 32 * 6  * 6;
    const size_t n3  = (size_t)B * 64 * 4  * 4;
    const size_t nf1 = (size_t)B * 120;
    const size_t nf2 = (size_t)B * 84;

    _Float16* H1 = (_Float16*)alloc(p1n * 2);   // largest H1 resident: pooled1
    _Float16* H2 = (_Float16*)alloc(n1 * 2);    // largest H2 resident: q1
    float*    F  = (float*)alloc(n1 * 4);       // largest f32 pre-quant: y1

    const dim3 blk(256);
    auto gemm = [&](const _Float16* A, const _Float16* Wq, const float* bias,
                    float* Y, unsigned* omax, const unsigned* am, const unsigned* wm,
                    float aQ, float wQ, int M, int N, int Npad, int Kreal, int Kpad,
                    int conv, int Cin, int H, int W, int OH, int OW, int relu) {
        dim3 g((unsigned)((M + 63) / 64), (unsigned)((N + 31) / 32));
        k_gemm_wmma<<<g, blk, 0, stream>>>(A, Wq, bias, Y, omax, am, wm, aQ, wQ,
                                           M, N, Npad, Kreal, Kpad, conv, Cin,
                                           H, W, OH, OW, relu);
    };

    // reset per-tensor max accumulators (deterministic per call)
    k_zero16<<<1, 32, 0, stream>>>(sc);

    // input 4-bit signed fake-quant -> integer-valued f16
    k_absmax<<<2048, blk, 0, stream>>>(x, (long long)nx, sc + 0);
    k_quant_s4<<<2048, blk, 0, stream>>>(x, (long long)nx, sc + 0, H1);

    // weight 3-bit narrow fake-quant -> integer-valued f16, padded [Npad][Kpad]
    k_quant_w<<<1, blk, 0, stream>>>(w1,  16,  32,  27,   64,   sc + 1, w1q, 3.f);
    k_quant_w<<<1, blk, 0, stream>>>(w2,  32,  32,  144,  192,  sc + 2, w2q, 3.f);
    k_quant_w<<<1, blk, 0, stream>>>(w3,  64,  64,  288,  320,  sc + 3, w3q, 3.f);
    k_quant_w<<<1, blk, 0, stream>>>(fw1, 120, 128, 1024, 1024, sc + 4, f1q, 3.f);
    k_quant_w<<<1, blk, 0, stream>>>(fw2, 84,  96,  120,  128,  sc + 5, f2q, 3.f);
    k_quant_w<<<1, blk, 0, stream>>>(fw3, 10,  32,  84,   128,  sc + 6, f3q, 3.f);

    // conv1: [B,3,32,32] -> [B,16,30,30], relu + max
    gemm(H1, w1q, b1, F, sc + 7, sc + 0, sc + 1, 7.f, 3.f,
         B * 900, 16, 32, 27, 64, 1, 3, 32, 32, 30, 30, 1);
    k_quant_u4<<<2048, blk, 0, stream>>>(F, (long long)n1, sc + 7, H2);
    k_maxpool<<<(unsigned)((p1n + 255) / 256), blk, 0, stream>>>(H2, H1, B * 16, 30, 30, 15, 15);

    // conv2: [B,16,15,15] -> [B,32,13,13]
    gemm(H1, w2q, b2, F, sc + 8, sc + 7, sc + 2, 15.f, 3.f,
         B * 169, 32, 32, 144, 192, 1, 16, 15, 15, 13, 13, 1);
    k_quant_u4<<<2048, blk, 0, stream>>>(F, (long long)n2, sc + 8, H2);
    k_maxpool<<<(unsigned)((p2n + 255) / 256), blk, 0, stream>>>(H2, H1, B * 32, 13, 13, 6, 6);

    // conv3: [B,32,6,6] -> [B,64,4,4]
    gemm(H1, w3q, b3, F, sc + 9, sc + 8, sc + 3, 15.f, 3.f,
         B * 16, 64, 64, 288, 320, 1, 32, 6, 6, 4, 4, 1);
    k_quant_u4<<<2048, blk, 0, stream>>>(F, (long long)n3, sc + 9, H2);  // = [B,1024] flat

    // fc1: 1024 -> 120
    gemm(H2, f1q, fb1, F, sc + 10, sc + 9, sc + 4, 15.f, 3.f,
         B, 120, 128, 1024, 1024, 0, 0, 0, 0, 1, 1, 1);
    k_quant_u4<<<2048, blk, 0, stream>>>(F, (long long)nf1, sc + 10, H1);

    // fc2: 120 -> 84
    gemm(H1, f2q, fb2, F, sc + 11, sc + 10, sc + 5, 15.f, 3.f,
         B, 84, 96, 120, 128, 0, 0, 0, 0, 1, 1, 1);
    k_quant_u4<<<2048, blk, 0, stream>>>(F, (long long)nf2, sc + 11, H2);

    // fc3: 84 -> 10, no bias / relu / quant, f32 out
    gemm(H2, f3q, nullptr, out, nullptr, sc + 11, sc + 6, 15.f, 3.f,
         B, 10, 32, 84, 128, 0, 0, 0, 0, 1, 1, 0);
}
